// SemanticConsistencyRegularizer_11570641895637
// MI455X (gfx1250) — matrix operand order repaired and checked by reading, not compiled, not run on hardware
//
#include <hip/hip_runtime.h>

#define B_N 8192
#define D_K 256
#define LAMBDA_REG 0.1f

typedef __attribute__((ext_vector_type(16))) _Float16 v16h;
typedef __attribute__((ext_vector_type(8)))  _Float16 v8h;
typedef __attribute__((ext_vector_type(8)))  float    v8f;

union FragAB { v16h v; v8h h[2]; };
union AccC   { v8f  v; float f[8]; };

// ---------------------------------------------------------------------------
// Kernel 1: row-normalize fp32 embeddings -> f16 normalized matrix in ws.
// One 256-thread block per row (8 waves); wave shuffle + LDS reduction.
// ---------------------------------------------------------------------------
__global__ void normalize_rows_kernel(const float* __restrict__ emb,
                                      _Float16* __restrict__ en) {
  const int row = blockIdx.x;
  const int tid = threadIdx.x;
  const float x = emb[(size_t)row * D_K + tid];

  float sq = x * x;
  #pragma unroll
  for (int off = 16; off > 0; off >>= 1)
    sq += __shfl_down(sq, off, 32);

  __shared__ float s[8];
  if ((tid & 31) == 0) s[tid >> 5] = sq;
  __syncthreads();

  float tot = 0.0f;
  #pragma unroll
  for (int w = 0; w < 8; ++w) tot += s[w];

  const float inv = 1.0f / fmaxf(sqrtf(tot), 1e-8f);
  en[(size_t)row * D_K + tid] = (_Float16)(x * inv);
}

// ---------------------------------------------------------------------------
// Kernel 2: fused Gram-matrix + masked |text_sim - cos| reduction.
// Block = 128 threads = 4 waves; each wave computes one 16x16 tile of a
// 32x32 region via 8x v_wmma_f32_16x16x32_f16. Tiles strictly below the
// diagonal are skipped wave-uniformly (EXEC stays all-ones for WMMA).
// text_sim is streamed with non-temporal loads (read-once data; keep the
// L2 for the hot 4MB f16 operand matrix). Branchless i<j masking in the
// epilogue. One float partial per block (deterministic).
// ---------------------------------------------------------------------------
__global__ void cos_loss_tiles_kernel(const _Float16* __restrict__ en,
                                      const float* __restrict__ ts,
                                      float* __restrict__ partials) {
  const int wave = threadIdx.x >> 5;
  const int lane = threadIdx.x & 31;
  const int half = lane >> 4;   // 0: lanes 0-15, 1: lanes 16-31
  const int l16  = lane & 15;

  const int i0 = blockIdx.y * 32 + (wave >> 1) * 16;
  const int j0 = blockIdx.x * 32 + (wave & 1) * 16;

  float p = 0.0f;

  if (j0 >= i0) {  // tile intersects the strict upper triangle
    AccC c;
    #pragma unroll
    for (int r = 0; r < 8; ++r) c.f[r] = 0.0f;

    const _Float16* arow = en + (size_t)(i0 + l16) * D_K;  // A row (M = l16)
    const _Float16* brow = en + (size_t)(j0 + l16) * D_K;  // B col (N = l16)

    #pragma unroll
    for (int k0 = 0; k0 < D_K; k0 += 32) {
      FragAB a, b;
      // 16-bit A 16x32 layout: lanes 0-15 hold K {0..7, 16..23},
      //                        lanes 16-31 hold K {8..15, 24..31}
      a.h[0] = *(const v8h*)(arow + k0 + half * 8);
      a.h[1] = *(const v8h*)(arow + k0 + 16 + half * 8);
      // 16-bit B 32x16 layout: lanes 0-15 hold K {0..7, 16..23},
      //                        lanes 16-31 hold K {8..15, 24..31}
      b.h[0] = *(const v8h*)(brow + k0 + half * 8);
      b.h[1] = *(const v8h*)(brow + k0 + 16 + half * 8);

      c.v = __builtin_amdgcn_wmma_f32_16x16x32_f16(
          /*neg_a=*/false, a.v, /*neg_b=*/false, b.v,
          /*c_mod=*/(short)0, c.v, /*reuse_a=*/false, /*reuse_b=*/false);
    }

    // C layout: VGPR r -> M = r + 8*half (lanes 0-15: M=r, lanes 16-31: M=r+8),
    // N = lane & 15.
    const int j = j0 + l16;
    const float* tsrow = ts + (size_t)(i0 + half * 8) * B_N + j;
    #pragma unroll
    for (int r = 0; r < 8; ++r) {
      const int i = i0 + r + half * 8;
      // Unconditional streaming (non-temporal) load; in-bounds for the full
      // [B,B] matrix. Branchless mask instead of exec-divergent branch.
      const float tv = __builtin_nontemporal_load(tsrow + (size_t)r * B_N);
      const float m  = (i < j) ? 1.0f : 0.0f;
      p += m * fabsf(tv - c.f[r]);
    }
  }

  // Wave reduce (wave32), then block reduce via LDS.
  #pragma unroll
  for (int off = 16; off > 0; off >>= 1)
    p += __shfl_down(p, off, 32);

  __shared__ float s[4];
  if (lane == 0) s[wave] = p;
  __syncthreads();
  if (threadIdx.x == 0)
    partials[(size_t)blockIdx.y * gridDim.x + blockIdx.x] =
        s[0] + s[1] + s[2] + s[3];
}

// ---------------------------------------------------------------------------
// Kernel 3: deterministic tree reduction of block partials + final scaling.
// ---------------------------------------------------------------------------
__global__ void reduce_partials_kernel(const float* __restrict__ partials,
                                       int n, float* __restrict__ out) {
  __shared__ float s[256];
  float p = 0.0f;
  for (int idx = threadIdx.x; idx < n; idx += 256) p += partials[idx];
  s[threadIdx.x] = p;
  __syncthreads();
  #pragma unroll
  for (int off = 128; off > 0; off >>= 1) {
    if (threadIdx.x < off) s[threadIdx.x] += s[threadIdx.x + off];
    __syncthreads();
  }
  if (threadIdx.x == 0) {
    const float count = (float)B_N * (float)(B_N - 1) * 0.5f;
    out[0] = LAMBDA_REG * s[0] / count;
  }
}

// ---------------------------------------------------------------------------
extern "C" void kernel_launch(void* const* d_in, const int* in_sizes, int n_in,
                              void* d_out, int out_size, void* d_ws, size_t ws_size,
                              hipStream_t stream) {
  (void)in_sizes; (void)n_in; (void)out_size; (void)ws_size;

  const float* emb = (const float*)d_in[0];   // [8192, 256] fp32
  const float* ts  = (const float*)d_in[1];   // [8192, 8192] fp32
  float* out = (float*)d_out;                 // scalar fp32

  // Workspace layout: [0, 4MB) f16 normalized matrix; then block partials.
  _Float16* en = (_Float16*)d_ws;
  float* partials =
      (float*)((char*)d_ws + (size_t)B_N * D_K * sizeof(_Float16));

  normalize_rows_kernel<<<B_N, 256, 0, stream>>>(emb, en);

  dim3 grid(B_N / 32, B_N / 32);  // 256 x 256 blocks of 32x32 tiles
  cos_loss_tiles_kernel<<<grid, 128, 0, stream>>>(en, ts, partials);

  const int n_partials = (B_N / 32) * (B_N / 32);
  reduce_partials_kernel<<<1, 256, 0, stream>>>(partials, n_partials, out);
}
